// SparseMoETransformer_25074019074701
// MI455X (gfx1250) — matrix operand order repaired and checked by reading, not compiled, not run on hardware
//
#include <hip/hip_runtime.h>
#include <math.h>

// ---------------------------------------------------------------------------
// CDNA5 / gfx1250: wave32, WMMA f32_16x16x32_f16, 320KB LDS per WGP.
// SparseMoE transformer forward. Only layer L-1 affects the output
// (reference overwrites `attens` from raw embeddings each iteration).
// GEMM: software-pipelined + double-buffered LDS (1 barrier / K-step);
// global loads for tile k+1 issue before the WMMA block of tile k, so
// HBM latency hides behind matrix-core work. 32x64 per-wave tile:
// 8 WMMAs per 12 ds_load_b128 (B-fragment reuse across two A-fragments).
// ---------------------------------------------------------------------------

typedef __attribute__((ext_vector_type(16))) _Float16 v16h;
typedef __attribute__((ext_vector_type(8)))  _Float16 v8h;
typedef __attribute__((ext_vector_type(4)))  _Float16 v4h;
typedef __attribute__((ext_vector_type(8)))  float    v8f;
typedef __attribute__((ext_vector_type(4)))  float    v4f;

constexpr int Bn = 8, Sn = 1024, Dn = 512, Hn = 8, En = 8, Vn = 8192, Ln = 2;
constexpr int HSn = Dn / Hn;        // 64
constexpr int FFn = 4 * Dn;         // 2048
constexpr int Mrows = Bn * Sn;      // 8192

static __device__ __forceinline__ v16h concat8(v8h lo, v8h hi) {
    return __builtin_shufflevector(lo, hi, 0, 1, 2, 3, 4, 5, 6, 7,
                                           8, 9, 10, 11, 12, 13, 14, 15);
}

// ---------------------------------------------------------------------------
// WMMA GEMM:  C[M,N] = epi( A[M,K](f32) * Wt[N,K](f16, pre-transposed) )
// epi: +bias, optional relu, optional per-row scale (gates), optional add
// (residual), optional accumulate into C.
// Workgroup: 256 threads = 8 waves as 2(M) x 4(N); WG tile 64x256,
// wave tile 32x64; K stepped by 32, software-pipelined double buffer.
// ---------------------------------------------------------------------------
__global__ __launch_bounds__(256)
void gemm_wmma_f16(const float* __restrict__ A, const _Float16* __restrict__ Wt,
                   const float* __restrict__ bias, const float* __restrict__ add,
                   const float* __restrict__ rowscale, int rs_stride,
                   float* __restrict__ C, int M, int N, int Kd,
                   int relu_act, int accum)
{
    __shared__ _Float16 As[2][64][40];      // 64 rows x 32 K, stride 80B
    __shared__ _Float16 Ws[2][256][40];     // 256 cols x 32 K (column-major)

    const int tid  = threadIdx.x;
    const int lane = tid & 31;
    const int w    = tid >> 5;
    const int wm   = w & 1;                 // M sub-tile (0..1): 32 rows
    const int wn   = w >> 1;                // N sub-tile (0..3): 64 cols
    const int m0   = blockIdx.y * 64;
    const int n0   = blockIdx.x * 256;
    const int g    = lane >> 4;             // lane half (wave32)
    const int ln15 = lane & 15;

    // Staging coordinates (all vectorized)
    const int ar = tid >> 2;                // A: 64 rows, 4 threads/row
    const int ac = (tid & 3) * 8;           //    8 f32 each
    // W: one 32-half column per thread (tid = column index in tile)

    // Software-pipeline staging registers (global -> regs -> LDS)
    v4f ra0, ra1;
    v8h rw0, rw1, rw2, rw3;

    auto loadStage = [&](int k0) {
        const float* srcA = &A[(size_t)(m0 + ar) * Kd + k0 + ac];
        ra0 = *(const v4f*)(srcA);
        ra1 = *(const v4f*)(srcA + 4);
        const _Float16* srcW = &Wt[(size_t)(n0 + tid) * Kd + k0];
        rw0 = *(const v8h*)(srcW);
        rw1 = *(const v8h*)(srcW + 8);
        rw2 = *(const v8h*)(srcW + 16);
        rw3 = *(const v8h*)(srcW + 24);
    };
    auto storeStage = [&](int buf) {
        v8h hv;
        #pragma unroll
        for (int i = 0; i < 4; ++i) { hv[i] = (_Float16)ra0[i]; hv[i + 4] = (_Float16)ra1[i]; }
        *(v8h*)&As[buf][ar][ac] = hv;
        *(v8h*)&Ws[buf][tid][0]  = rw0;
        *(v8h*)&Ws[buf][tid][8]  = rw1;
        *(v8h*)&Ws[buf][tid][16] = rw2;
        *(v8h*)&Ws[buf][tid][24] = rw3;
    };

    v8f acc[2][4] = {};                     // 32x64 f32 accumulators per wave

    const int nk = Kd >> 5;
    loadStage(0);
    for (int kt = 0; kt < nk; ++kt) {
        const int cur = kt & 1;
        storeStage(cur);
        __syncthreads();                    // tile cur visible; prev reads done
        if (kt + 1 < nk) {
            loadStage((kt + 1) * 32);       // issue early: hides behind WMMAs
            // Prefetch tile kt+2 weights into L2 (global_prefetch_b8)
            if (kt + 2 < nk)
                __builtin_prefetch(&Wt[(size_t)(n0 + tid) * Kd + (kt + 2) * 32], 0, 1);
        }

        // A fragments 16x32: lanes 0-15 hold K 0-7/16-23, 16-31 K 8-15/24-31
        v16h a[2];
        #pragma unroll
        for (int r2 = 0; r2 < 2; ++r2) {
            const _Float16* ap = &As[cur][wm * 32 + r2 * 16 + ln15][8 * g];
            a[r2] = concat8(*(const v8h*)ap, *(const v8h*)(ap + 16));
        }
        #pragma unroll
        for (int c = 0; c < 4; ++c) {
            // B fragment 32x16: N = lane&15, K 0-7/16-23 (half 0), 8-15/24-31 (half 1)
            const _Float16* bp = &Ws[cur][wn * 64 + c * 16 + ln15][8 * g];
            v16h b = concat8(*(const v8h*)bp, *(const v8h*)(bp + 16));
            #pragma unroll
            for (int r2 = 0; r2 < 2; ++r2)
                acc[r2][c] = __builtin_amdgcn_wmma_f32_16x16x32_f16(
                    false, a[r2], false, b, (short)0, acc[r2][c], false, false);
        }
    }

    // Epilogue.  C/D layout: M = j + 8*half, N = lane&15.
    #pragma unroll
    for (int r2 = 0; r2 < 2; ++r2) {
        #pragma unroll
        for (int c = 0; c < 4; ++c) {
            #pragma unroll
            for (int j = 0; j < 8; ++j) {
                int row = m0 + wm * 32 + r2 * 16 + j + 8 * g;
                int col = n0 + wn * 64 + c * 16 + ln15;
                float v = acc[r2][c][j];
                if (bias)     v += bias[col];
                if (relu_act) v = v > 0.f ? v : 0.f;
                if (rowscale) v *= rowscale[(size_t)row * rs_stride];
                if (add)      v += add[(size_t)row * N + col];
                size_t o = (size_t)row * N + col;
                C[o] = accum ? (C[o] + v) : v;
            }
        }
    }
}

// ---------------------------------------------------------------------------
// Flash attention per (b,h): scores = (q/sqrt(D)) k^T, causal softmax, P*V.
// 8 waves x 16 query rows = 128 rows / workgroup; 32-key tiles in LDS.
// Ks kept [key][hs] (contiguous hs-run for score B-frags); V staged
// transposed as Vt[hs][key] (contiguous key-run for PV B-frags).
// ---------------------------------------------------------------------------
__global__ __launch_bounds__(256)
void attn_flash(const float* __restrict__ Q, const float* __restrict__ Kb,
                const float* __restrict__ Vb, float* __restrict__ O)
{
    __shared__ _Float16 Ks[32][72];      // [key][hs], stride 144B
    __shared__ _Float16 Vt[64][40];      // [hs][key], stride 80B
    __shared__ _Float16 Ps[8][16][40];   // per-wave P scratch (C->A relayout)

    const int tid  = threadIdx.x;
    const int lane = tid & 31;
    const int w    = tid >> 5;
    const int b    = blockIdx.x / Hn;
    const int h    = blockIdx.x % Hn;
    const int q0   = blockIdx.y * 128 + w * 16;
    const int g    = lane >> 4;
    const int ln15 = lane & 15;
    const float scale = rsqrtf((float)Dn);   // reference divides by sqrt(embed)

    // q fragments: two 16x32 chunks across HS=64, vector loads, pre-scaled
    v16h qa[2];
    #pragma unroll
    for (int u = 0; u < 2; ++u) {
        const float* src = &Q[((size_t)(b * Sn) + q0 + ln15) * Dn + h * HSn + u * 32 + 8 * g];
        v4f x0 = *(const v4f*)(src);
        v4f x1 = *(const v4f*)(src + 4);
        v4f x2 = *(const v4f*)(src + 16);
        v4f x3 = *(const v4f*)(src + 20);
        #pragma unroll
        for (int i = 0; i < 4; ++i) {
            qa[u][i]      = (_Float16)(x0[i] * scale);
            qa[u][i + 4]  = (_Float16)(x1[i] * scale);
            qa[u][i + 8]  = (_Float16)(x2[i] * scale);
            qa[u][i + 12] = (_Float16)(x3[i] * scale);
        }
    }

    v8f o[4] = {};
    float mrow[8], lrow[8];
    #pragma unroll
    for (int j = 0; j < 8; ++j) { mrow[j] = -1e30f; lrow[j] = 0.f; }

    const int ntiles = blockIdx.y * 4 + 4;   // keys up to WG's max query row
    for (int t = 0; t < ntiles; ++t) {
        const int kbase = t * 32;
        // Stage K (direct) and V (transposed), 32x64 each, f32 -> f16
        #pragma unroll
        for (int it = 0; it < 2; ++it) {
            int idx = tid + it * 256;            // 512 chunks of 4
            int kk = idx >> 4, c0 = (idx & 15) * 4;
            size_t src = ((size_t)(b * Sn) + kbase + kk) * Dn + h * HSn + c0;
            v4f kv = *(const v4f*)&Kb[src];
            v4f vv = *(const v4f*)&Vb[src];
            v4h kh;
            #pragma unroll
            for (int j = 0; j < 4; ++j) kh[j] = (_Float16)kv[j];
            *(v4h*)&Ks[kk][c0] = kh;
            #pragma unroll
            for (int j = 0; j < 4; ++j) Vt[c0 + j][kk] = (_Float16)vv[j];
        }
        __syncthreads();

        // scores 16x32 via chained WMMAs over HS
        v8f s[2] = {};
        #pragma unroll
        for (int c = 0; c < 2; ++c) {
            #pragma unroll
            for (int u = 0; u < 2; ++u) {
                // B = K^T: N = key (lane&15), K-dim = hs = u*32 + {0-7,16-23} + 8*half
                const _Float16* kp = &Ks[c * 16 + ln15][u * 32 + 8 * g];
                v16h bk = concat8(*(const v8h*)kp, *(const v8h*)(kp + 16));
                s[c] = __builtin_amdgcn_wmma_f32_16x16x32_f16(
                    false, qa[u], false, bk, (short)0, s[c], false, false);
            }
        }

        // causal mask + online softmax (row = j + 8*half lives in one 16-lane half)
        #pragma unroll
        for (int j = 0; j < 8; ++j) {
            int qrow = q0 + j + 8 * g;
            #pragma unroll
            for (int c = 0; c < 2; ++c) {
                int key = kbase + c * 16 + ln15;
                if (key > qrow) s[c][j] = -1e30f;
            }
            float mx = fmaxf(s[0][j], s[1][j]);
            #pragma unroll
            for (int off = 1; off < 16; off <<= 1) mx = fmaxf(mx, __shfl_xor(mx, off));
            float mnew = fmaxf(mrow[j], mx);
            float corr = __expf(mrow[j] - mnew);
            float psum = 0.f;
            #pragma unroll
            for (int c = 0; c < 2; ++c) {
                float p = __expf(s[c][j] - mnew);
                s[c][j] = p; psum += p;
            }
            #pragma unroll
            for (int off = 1; off < 16; off <<= 1) psum += __shfl_xor(psum, off);
            lrow[j] = lrow[j] * corr + psum;
            mrow[j] = mnew;
            #pragma unroll
            for (int c4 = 0; c4 < 4; ++c4) o[c4][j] *= corr;
        }

        // Relayout P (C layout) -> A fragment via per-wave LDS
        #pragma unroll
        for (int j = 0; j < 8; ++j)
            #pragma unroll
            for (int c = 0; c < 2; ++c)
                Ps[w][j + 8 * g][c * 16 + ln15] = (_Float16)s[c][j];
        asm volatile("s_wait_dscnt 0" ::: "memory");   // same-wave LDS RAW
        v8h plo = *(const v8h*)&Ps[w][ln15][8 * g];
        v8h phi = *(const v8h*)&Ps[w][ln15][8 * g + 16];
        v16h pa = concat8(plo, phi);

        // o += P(16x32) * V(32x64): B from Vt, N = hs, K = key = {0-7,16-23} + 8*half
        #pragma unroll
        for (int c4 = 0; c4 < 4; ++c4) {
            const _Float16* vp = &Vt[c4 * 16 + ln15][8 * g];
            v16h bv = concat8(*(const v8h*)vp, *(const v8h*)(vp + 16));
            o[c4] = __builtin_amdgcn_wmma_f32_16x16x32_f16(
                false, pa, false, bv, (short)0, o[c4], false, false);
        }
        __syncthreads();
    }

    // normalize + store concat-heads layout [B,S,D]
    #pragma unroll
    for (int j = 0; j < 8; ++j) {
        float inv = 1.0f / lrow[j];
        int row = q0 + j + 8 * g;
        #pragma unroll
        for (int c4 = 0; c4 < 4; ++c4)
            O[((size_t)(b * Sn) + row) * Dn + h * HSn + c4 * 16 + ln15] = o[c4][j] * inv;
    }
}

// ---------------------------------------------------------------------------
// LayerNorm over D=512: one wave per row (wave32: 16 elems/lane).
// ---------------------------------------------------------------------------
__global__ __launch_bounds__(256)
void layernorm_k(const float* __restrict__ X, const float* __restrict__ gw,
                 const float* __restrict__ bw, float* __restrict__ Y, int Mr)
{
    int row = blockIdx.x * (blockDim.x >> 5) + (threadIdx.x >> 5);
    if (row >= Mr) return;
    int lane = threadIdx.x & 31;
    const float* x = X + (size_t)row * Dn;
    float s = 0.f, s2 = 0.f;
    #pragma unroll
    for (int i = 0; i < 16; ++i) { float v = x[lane + i * 32]; s += v; s2 += v * v; }
    #pragma unroll
    for (int off = 1; off < 32; off <<= 1) { s += __shfl_xor(s, off); s2 += __shfl_xor(s2, off); }
    float mean = s * (1.0f / Dn);
    float var  = s2 * (1.0f / Dn) - mean * mean;
    float r = rsqrtf(var + 1e-5f);
    #pragma unroll
    for (int i = 0; i < 16; ++i) {
        int c = lane + i * 32;
        Y[(size_t)row * Dn + c] = (x[c] - mean) * r * gw[c] + bw[c];
    }
}

// x = tok[idx] + pos
__global__ void embed_k(const int* __restrict__ idx, const float* __restrict__ tok,
                        const float* __restrict__ pos, float* __restrict__ X)
{
    size_t i = (size_t)blockIdx.x * blockDim.x + threadIdx.x;   // Mrows*Dn
    int m = (int)(i >> 9), d = (int)(i & 511);
    int s = m % Sn;
    X[i] = tok[(size_t)idx[m] * Dn + d] + pos[(size_t)s * Dn + d];
}

// router head: scores[m,e] = RH[m,:] . Rw2[:,e] + Rb2[e]   (N=8, VALU)
__global__ void router2_k(const float* __restrict__ RH, const float* __restrict__ Rw2,
                          const float* __restrict__ Rb2, float* __restrict__ SC)
{
    int i = blockIdx.x * blockDim.x + threadIdx.x;
    if (i >= Mrows * En) return;
    int m = i >> 3, e = i & 7;
    float acc = Rb2[e];
    const float* rh = RH + (size_t)m * FFn;
    for (int f = 0; f < FFn; ++f) acc += rh[f] * Rw2[(size_t)f * En + e];
    SC[i] = acc;
}

// top-2 + softmax over selected -> gates (0 elsewhere)
__global__ void topk_gate_k(const float* __restrict__ SC, float* __restrict__ GT)
{
    int m = blockIdx.x * blockDim.x + threadIdx.x;
    if (m >= Mrows) return;
    const float* s = SC + (size_t)m * En;
    int i1 = 0; float v1 = s[0];
    for (int e = 1; e < En; ++e) if (s[e] > v1) { v1 = s[e]; i1 = e; }
    int i2 = (i1 == 0) ? 1 : 0; float v2 = s[i2];
    for (int e = 0; e < En; ++e) if (e != i1 && s[e] > v2) { v2 = s[e]; i2 = e; }
    float e2 = __expf(v2 - v1);
    float inv = 1.0f / (1.0f + e2);
    #pragma unroll
    for (int e = 0; e < En; ++e) GT[(size_t)m * En + e] = 0.f;
    GT[(size_t)m * En + i1] = inv;
    GT[(size_t)m * En + i2] = e2 * inv;
}

// Transpose-pack: src f32 [cnt][K][N] -> dst f16 [cnt][N][K]
__global__ void packT_k(const float* __restrict__ src, _Float16* __restrict__ dst,
                        int Kd, int N, size_t total)
{
    size_t i = (size_t)blockIdx.x * blockDim.x + threadIdx.x;
    if (i >= total) return;
    size_t per = (size_t)Kd * N;
    size_t mtx = i / per, r = i % per;
    int n = (int)(r / Kd), k = (int)(r % Kd);
    dst[i] = (_Float16)src[mtx * per + (size_t)k * N + n];
}

// Wq[h][d][hs] -> Wt[n][d] (n = h*64+hs), f16 transposed for GEMM
__global__ void pack_qkvT_k(const float* __restrict__ src, _Float16* __restrict__ dst)
{
    int i = blockIdx.x * blockDim.x + threadIdx.x;      // Dn*Dn, dst-linear
    if (i >= Dn * Dn) return;
    int n = i >> 9, d = i & 511;
    int h = n >> 6, k = n & 63;
    dst[i] = (_Float16)src[((size_t)h * Dn + d) * HSn + k];
}

// ---------------------------------------------------------------------------
extern "C" void kernel_launch(void* const* d_in, const int* in_sizes, int n_in,
                              void* d_out, int out_size, void* d_ws, size_t ws_size,
                              hipStream_t stream)
{
    (void)in_sizes; (void)n_in; (void)out_size; (void)ws_size;
    const int l = Ln - 1;   // only the last layer reaches the output

    const int*   inputs = (const int*)d_in[0];
    const float* tok    = (const float*)d_in[1];
    const float* pos    = (const float*)d_in[2];
    const float* ln1_g  = (const float*)d_in[3] + (size_t)l * Dn;
    const float* ln1_b  = (const float*)d_in[4] + (size_t)l * Dn;
    const float* Wq     = (const float*)d_in[5] + (size_t)l * Hn * Dn * HSn;
    const float* bq     = (const float*)d_in[6] + (size_t)l * Dn;   // (H,HS) contig = 512
    const float* Wk     = (const float*)d_in[7] + (size_t)l * Hn * Dn * HSn;
    const float* bk     = (const float*)d_in[8] + (size_t)l * Dn;
    const float* Wv     = (const float*)d_in[9] + (size_t)l * Hn * Dn * HSn;
    const float* bv     = (const float*)d_in[10] + (size_t)l * Dn;
    const float* Wo     = (const float*)d_in[11] + (size_t)l * Dn * Dn;
    const float* bo     = (const float*)d_in[12] + (size_t)l * Dn;
    const float* ln2_g  = (const float*)d_in[13] + (size_t)l * Dn;
    const float* ln2_b  = (const float*)d_in[14] + (size_t)l * Dn;
    const float* Rw1    = (const float*)d_in[15] + (size_t)l * Dn * FFn;
    const float* Rb1    = (const float*)d_in[16] + (size_t)l * FFn;
    const float* Rw2    = (const float*)d_in[17] + (size_t)l * FFn * En;
    const float* Rb2    = (const float*)d_in[18] + (size_t)l * En;
    const float* Ew1    = (const float*)d_in[19] + (size_t)l * En * Dn * FFn;
    const float* Eb1    = (const float*)d_in[20] + (size_t)l * En * FFn;
    const float* Ew2    = (const float*)d_in[21] + (size_t)l * En * FFn * Dn;
    const float* Eb2    = (const float*)d_in[22] + (size_t)l * En * Dn;
    const float* lnf_g  = (const float*)d_in[23];
    const float* lnf_b  = (const float*)d_in[24];
    const float* Wout   = (const float*)d_in[25];
    const float* bout   = (const float*)d_in[26];
    float* out = (float*)d_out;

    // ---- workspace bump allocator ----
    char* ws = (char*)d_ws;
    size_t off = 0;
    auto alloc = [&](size_t bytes) -> void* {
        void* p = ws + off; off += (bytes + 255) & ~(size_t)255; return p;
    };
    float* X   = (float*)alloc((size_t)Mrows * Dn * 4);
    float* Y   = (float*)alloc((size_t)Mrows * Dn * 4);
    float* QB  = (float*)alloc((size_t)Mrows * Dn * 4);
    float* KB  = (float*)alloc((size_t)Mrows * Dn * 4);
    float* VB  = (float*)alloc((size_t)Mrows * Dn * 4);
    float* OB  = (float*)alloc((size_t)Mrows * Dn * 4);
    float* Y2  = (float*)alloc((size_t)Mrows * Dn * 4);
    float* Z   = (float*)alloc((size_t)Mrows * Dn * 4);
    float* MOE = (float*)alloc((size_t)Mrows * Dn * 4);
    float* HF  = (float*)alloc((size_t)Mrows * Dn * 4);
    float* RH  = (float*)alloc((size_t)Mrows * FFn * 4);   // router hidden, reused as expert hidden
    float* SC  = (float*)alloc((size_t)Mrows * En * 4);
    float* GT  = (float*)alloc((size_t)Mrows * En * 4);
    _Float16* WqP   = (_Float16*)alloc((size_t)Dn * Dn * 2);     // [N][K]
    _Float16* WkP   = (_Float16*)alloc((size_t)Dn * Dn * 2);
    _Float16* WvP   = (_Float16*)alloc((size_t)Dn * Dn * 2);
    _Float16* WoP   = (_Float16*)alloc((size_t)Dn * Dn * 2);
    _Float16* Rw1P  = (_Float16*)alloc((size_t)FFn * Dn * 2);
    _Float16* Ew1P  = (_Float16*)alloc((size_t)En * FFn * Dn * 2);
    _Float16* Ew2P  = (_Float16*)alloc((size_t)En * Dn * FFn * 2);
    _Float16* WoutP = (_Float16*)alloc((size_t)Vn * Dn * 2);

    auto gemm = [&](const float* A, const _Float16* W, const float* bias,
                    const float* add, const float* rowscale, int rs_stride,
                    float* C, int M, int N, int K, int act, int accum) {
        dim3 grid(N / 256, M / 64);
        gemm_wmma_f16<<<grid, 256, 0, stream>>>(A, W, bias, add, rowscale, rs_stride,
                                                C, M, N, K, act, accum);
    };
    auto packT = [&](const float* s, _Float16* d, int K, int N, int cnt) {
        size_t total = (size_t)cnt * K * N;
        packT_k<<<(unsigned)((total + 255) / 256), 256, 0, stream>>>(s, d, K, N, total);
    };

    // ---- weight repack (f32 [K][N] -> f16 [N][K]) ----
    pack_qkvT_k<<<(Dn * Dn + 255) / 256, 256, 0, stream>>>(Wq, WqP);
    pack_qkvT_k<<<(Dn * Dn + 255) / 256, 256, 0, stream>>>(Wk, WkP);
    pack_qkvT_k<<<(Dn * Dn + 255) / 256, 256, 0, stream>>>(Wv, WvP);
    packT(Wo,   WoP,   Dn, Dn, 1);
    packT(Rw1,  Rw1P,  Dn, FFn, 1);
    packT(Ew1,  Ew1P,  Dn, FFn, En);
    packT(Ew2,  Ew2P,  FFn, Dn, En);
    packT(Wout, WoutP, Dn, Vn, 1);

    // ---- forward ----
    embed_k<<<(Mrows * Dn) / 256, 256, 0, stream>>>(inputs, tok, pos, X);
    layernorm_k<<<Mrows / 8, 256, 0, stream>>>(X, ln1_g, ln1_b, Y, Mrows);

    gemm(Y, WqP, bq, nullptr, nullptr, 0, QB, Mrows, Dn, Dn, 0, 0);
    gemm(Y, WkP, bk, nullptr, nullptr, 0, KB, Mrows, Dn, Dn, 0, 0);
    gemm(Y, WvP, bv, nullptr, nullptr, 0, VB, Mrows, Dn, Dn, 0, 0);

    attn_flash<<<dim3(Bn * Hn, Sn / 128), 256, 0, stream>>>(QB, KB, VB, OB);

    // y2 = O @ Wo + bo + y
    gemm(OB, WoP, bo, Y, nullptr, 0, Y2, Mrows, Dn, Dn, 0, 0);
    layernorm_k<<<Mrows / 8, 256, 0, stream>>>(Y2, ln2_g, ln2_b, Z, Mrows);

    // router
    gemm(Z, Rw1P, Rb1, nullptr, nullptr, 0, RH, Mrows, FFn, Dn, 1, 0);
    router2_k<<<(Mrows * En + 255) / 256, 256, 0, stream>>>(RH, Rw2, Rb2, SC);
    topk_gate_k<<<(Mrows + 255) / 256, 256, 0, stream>>>(SC, GT);

    // experts: moe = z + sum_e gate_e * (relu(z Ew1_e + b1) Ew2_e + b2)
    for (int e = 0; e < En; ++e) {
        gemm(Z, Ew1P + (size_t)e * FFn * Dn, Eb1 + (size_t)e * FFn,
             nullptr, nullptr, 0, RH, Mrows, FFn, Dn, 1, 0);
        gemm(RH, Ew2P + (size_t)e * Dn * FFn, Eb2 + (size_t)e * Dn,
             (e == 0) ? Z : nullptr, GT + e, En, MOE, Mrows, Dn, FFn, 0, (e == 0) ? 0 : 1);
    }

    // final LN + logits
    layernorm_k<<<Mrows / 8, 256, 0, stream>>>(MOE, lnf_g, lnf_b, HF, Mrows);
    gemm(HF, WoutP, bout, nullptr, nullptr, 0, out, Mrows, Vn, Dn, 0, 0);
}
